// SCAM_42039139893765
// MI455X (gfx1250) — matrix operand order repaired and checked by reading, not compiled, not run on hardware
//
#include <hip/hip_runtime.h>

typedef __attribute__((ext_vector_type(16))) _Float16 v16h;
typedef __attribute__((ext_vector_type(8)))  float    v8f;

#define CCH   64
#define HH    180
#define WW    320
#define BB    4
#define WHALF 160
#define NCH   16                 // 10-wide chunks per half-row
#define PLANE (HH*WW)            // 57600
#define IMG   (CCH*PLANE)        // 3686400
#define OUTR_OFS ((size_t)BB*IMG)

// workspace layout (floats)
#define WS_GAPL 0
#define WS_GAPR (WS_GAPL + BB*CCH)
#define WS_WAGL (WS_GAPR + BB*CCH)
#define WS_WAGR (WS_WAGL + BB*CCH*CCH)
#define WS_BAGL (WS_WAGR + BB*CCH*CCH)
#define WS_BAGR (WS_BAGL + BB*CCH)

// ---------------------------------------------------------------------------
// Pre-kernel 1: global average pool of x_l / x_r over (H,W) -> gap[B,C]
// ---------------------------------------------------------------------------
__global__ __launch_bounds__(256) void gap_kernel(const float* __restrict__ xl,
                                                  const float* __restrict__ xr,
                                                  float* __restrict__ ws) {
  __shared__ float red[256];
  const int idx  = blockIdx.x;        // 0..511
  const int side = idx >> 8;          // B*C = 256 per side
  const int bc   = idx & 255;
  const float* src = (side ? xr : xl) + (size_t)bc * PLANE;
  float s = 0.f;
  for (int i = threadIdx.x; i < PLANE; i += 256) s += src[i];
  red[threadIdx.x] = s;
  __syncthreads();
  for (int off = 128; off > 0; off >>= 1) {
    if (threadIdx.x < off) red[threadIdx.x] += red[threadIdx.x + off];
    __syncthreads();
  }
  if (threadIdx.x == 0)
    ws[(side ? WS_GAPR : WS_GAPL) + bc] = red[0] * (1.0f / PLANE);
}

// ---------------------------------------------------------------------------
// Pre-kernel 2: expert attention (K=3) + aggregated per-batch 64x64 weights
// ---------------------------------------------------------------------------
__global__ __launch_bounds__(64) void attw_kernel(
    const float* __restrict__ lf1w, const float* __restrict__ lf1b,
    const float* __restrict__ lf2w, const float* __restrict__ lf2b,
    const float* __restrict__ lfW,  const float* __restrict__ lfbias,
    const float* __restrict__ rf1w, const float* __restrict__ rf1b,
    const float* __restrict__ rf2w, const float* __restrict__ rf2b,
    const float* __restrict__ rfW,  const float* __restrict__ rfbias,
    float* __restrict__ ws) {
  const int side = blockIdx.x >> 2;
  const int b    = blockIdx.x & 3;
  const float* gap  = ws + (side ? WS_GAPR : WS_GAPL) + b * CCH;
  const float* fc1w = side ? rf1w : lf1w;
  const float* fc1b = side ? rf1b : lf1b;
  const float* fc2w = side ? rf2w : lf2w;
  const float* fc2b = side ? rf2b : lf2b;
  const float* Wexp = side ? rfW  : lfW;
  const float* bexp = side ? rfbias : lfbias;

  float a1[3], a2[3];
  #pragma unroll
  for (int k = 0; k < 3; ++k) {
    float s = fc1b[k];
    for (int c = 0; c < CCH; ++c) s += fc1w[k * CCH + c] * gap[c];
    a1[k] = fmaxf(s, 0.f);
  }
  #pragma unroll
  for (int k = 0; k < 3; ++k) {
    float s = fc2b[k];
    #pragma unroll
    for (int j = 0; j < 3; ++j) s += fc2w[k * 3 + j] * a1[j];
    a2[k] = s;
  }
  const float m  = fmaxf(a2[0], fmaxf(a2[1], a2[2]));
  const float e0 = __expf(a2[0] - m), e1 = __expf(a2[1] - m), e2 = __expf(a2[2] - m);
  const float rd = 1.f / (e0 + e1 + e2);
  const float t0 = e0 * rd, t1 = e1 * rd, t2 = e2 * rd;

  const int o = threadIdx.x;          // 0..63 -> one output channel
  float* wag = ws + (side ? WS_WAGR : WS_WAGL) + b * CCH * CCH;
  float* bag = ws + (side ? WS_BAGR : WS_BAGL) + b * CCH;
  for (int i = 0; i < CCH; ++i) {
    const int oi = o * CCH + i;
    wag[oi] = t0 * Wexp[oi] + t1 * Wexp[CCH * CCH + oi] + t2 * Wexp[2 * CCH * CCH + oi];
  }
  bag[o] = t0 * bexp[o] + t1 * bexp[CCH + o] + t2 * bexp[2 * CCH + o];
}

// ---------------------------------------------------------------------------
// WMMA helper: one 16x16 D tile of (64x64 A) x (B stored [w][c] f16), K=64
// A layout per ISA: lane holds M=lane%16, halves h -> K = h%8 + (h/8)*16 + (lane/16)*8
// B layout per ISA: lane holds N=lane%16, halves h -> K = h%8 + (h/8)*16 + (lane/16)*8
// ---------------------------------------------------------------------------
__device__ __forceinline__ v8f mm_tile(const _Float16* sA, const _Float16* sB,
                                       int mt, int nt, int lane, v8f acc) {
  const int lhi = lane >> 4;
  const int llo = lane & 15;
  #pragma unroll
  for (int kt = 0; kt < 2; ++kt) {
    v16h a, bmat;
    const _Float16* ar = sA + (mt * 16 + llo) * CCH + kt * 32 + lhi * 8;
    #pragma unroll
    for (int j = 0; j < 8; ++j) { a[j] = ar[j]; a[8 + j] = ar[16 + j]; }
    const _Float16* br = sB + (nt * 16 + llo) * CCH + kt * 32 + lhi * 8;
    #pragma unroll
    for (int j = 0; j < 8; ++j) { bmat[j] = br[j]; bmat[8 + j] = br[16 + j]; }
    acc = __builtin_amdgcn_wmma_f32_16x16x32_f16(false, a, false, bmat,
                                                 (short)0, acc, false, false);
  }
  return acc;
}

// ---------------------------------------------------------------------------
// Fused main kernel: one workgroup per (b, h, half-row of 160 columns)
// ---------------------------------------------------------------------------
__global__ __launch_bounds__(160) void fused_kernel(
    const float* __restrict__ xl, const float* __restrict__ xr,
    const float* __restrict__ nlw, const float* __restrict__ nlb,
    const float* __restrict__ nrw, const float* __restrict__ nrb,
    const float* __restrict__ lp1w, const float* __restrict__ lp1b,
    const float* __restrict__ rp1w, const float* __restrict__ rp1b,
    const float* __restrict__ lp2w, const float* __restrict__ lp2b,
    const float* __restrict__ rp2w, const float* __restrict__ rp2b,
    const float* __restrict__ beta, const float* __restrict__ gamma,
    const float* __restrict__ ws, float* __restrict__ out) {
  __shared__ _Float16 s_w[6][CCH * CCH];      // lp1,rp1,lp2,rp2,WaggL,WaggR (48KB)
  __shared__ float    s_vec[12][CCH];         // nlw,nlb,nrw,nrb,lp1b,rp1b,lp2b,rp2b,beta,gamma,bagL,bagR
  __shared__ _Float16 s_R1[2][WHALF * CCH];   // ping (20KB x2)
  __shared__ _Float16 s_R2[2][WHALF * CCH];   // pong (20KB x2)
  __shared__ float    s_att[NCH][10][10];     // raw attn -> A_l2r (in place)
  __shared__ float    s_ar [NCH][10][10];     // A_r2l

  const int tid  = threadIdx.x;               // 0..159, one width column
  const int bh   = blockIdx.x >> 1;
  const int wofs = (blockIdx.x & 1) * WHALF;
  const int b    = bh / HH;
  const int h    = bh % HH;

  // ---- stage 0: stage weights (f16) + vectors ----
  {
    const float* wagl = ws + WS_WAGL + b * CCH * CCH;
    const float* wagr = ws + WS_WAGR + b * CCH * CCH;
    for (int i = tid; i < CCH * CCH; i += 160) {
      s_w[0][i] = (_Float16)lp1w[i];
      s_w[1][i] = (_Float16)rp1w[i];
      s_w[2][i] = (_Float16)lp2w[i];
      s_w[3][i] = (_Float16)rp2w[i];
      s_w[4][i] = (_Float16)wagl[i];
      s_w[5][i] = (_Float16)wagr[i];
    }
    if (tid < CCH) {
      s_vec[0][tid] = nlw[tid];  s_vec[1][tid] = nlb[tid];
      s_vec[2][tid] = nrw[tid];  s_vec[3][tid] = nrb[tid];
      s_vec[4][tid] = lp1b[tid]; s_vec[5][tid] = rp1b[tid];
      s_vec[6][tid] = lp2b[tid]; s_vec[7][tid] = rp2b[tid];
      s_vec[8][tid] = beta[tid]; s_vec[9][tid] = gamma[tid];
      s_vec[10][tid] = ws[WS_BAGL + b * CCH + tid];
      s_vec[11][tid] = ws[WS_BAGR + b * CCH + tid];
    }
  }
  const float* rowl = xl + (size_t)b * IMG + (size_t)h * WW + wofs;
  const float* rowr = xr + (size_t)b * IMG + (size_t)h * WW + wofs;
  __syncthreads();

  // ---- stage 1: channel LayerNorm -> s_R1[side][w][c] (f16) ----
  {
    const int w = tid;
    float s0 = 0.f, q0 = 0.f, s1 = 0.f, q1 = 0.f;
    for (int c = 0; c < CCH; ++c) {
      const float vl = rowl[(size_t)c * PLANE + w];
      const float vr = rowr[(size_t)c * PLANE + w];
      s0 += vl; q0 += vl * vl; s1 += vr; q1 += vr * vr;
    }
    const float mu0 = s0 * (1.f / CCH), mu1 = s1 * (1.f / CCH);
    const float ri0 = rsqrtf(q0 * (1.f / CCH) - mu0 * mu0 + 1e-6f);
    const float ri1 = rsqrtf(q1 * (1.f / CCH) - mu1 * mu1 + 1e-6f);
    for (int c = 0; c < CCH; ++c) {
      const float vl = rowl[(size_t)c * PLANE + w];
      const float vr = rowr[(size_t)c * PLANE + w];
      s_R1[0][w * CCH + c] = (_Float16)(s_vec[0][c] * (vl - mu0) * ri0 + s_vec[1][c]);
      s_R1[1][w * CCH + c] = (_Float16)(s_vec[2][c] * (vr - mu1) * ri1 + s_vec[3][c]);
    }
  }
  __syncthreads();

  // ---- stage 2: Q = P1 @ LN(x) + b1  (WMMA) -> s_R2 ----
  {
    const int wave = tid >> 5, lane = tid & 31;
    const int lhi = lane >> 4, llo = lane & 15;
    for (int i = 0; i < 8; ++i) {
      const int t8 = wave * 8 + i;              // 0..39 tiles
      const int mt = t8 & 3, nt = t8 >> 2;
      #pragma unroll
      for (int side = 0; side < 2; ++side) {
        v8f acc;
        #pragma unroll
        for (int r = 0; r < 8; ++r) acc[r] = s_vec[4 + side][mt * 16 + lhi * 8 + r];
        acc = mm_tile(s_w[side], s_R1[side], mt, nt, lane, acc);
        #pragma unroll
        for (int r = 0; r < 8; ++r)
          s_R2[side][(nt * 16 + llo) * CCH + mt * 16 + lhi * 8 + r] = (_Float16)acc[r];
      }
    }
  }
  __syncthreads();

  // ---- stage 3: block attention (10x10 per chunk), both softmaxes ----
  {
    const int ch = tid / 10, u = tid % 10;
    const _Float16* qlrow = &s_R2[0][(ch * 10 + u) * CCH];
    for (int v = 0; v < 10; ++v) {
      const _Float16* qrrow = &s_R2[1][(ch * 10 + v) * CCH];
      float s = 0.f;
      for (int c = 0; c < CCH; ++c) s += (float)qlrow[c] * (float)qrrow[c];
      s_att[ch][u][v] = s * 0.125f;             // * C^-0.5
    }
  }
  __syncthreads();
  {                                             // A_r2l: softmax over v (rows)
    const int ch = tid / 10, u = tid % 10;
    float m = -1e30f, e[10], d = 0.f;
    for (int v = 0; v < 10; ++v) m = fmaxf(m, s_att[ch][u][v]);
    for (int v = 0; v < 10; ++v) { e[v] = __expf(s_att[ch][u][v] - m); d += e[v]; }
    const float rd = 1.f / d;
    for (int v = 0; v < 10; ++v) s_ar[ch][u][v] = e[v] * rd;
  }
  __syncthreads();
  {                                             // A_l2r: softmax over u (cols, in place)
    const int ch = tid / 10, v = tid % 10;
    float m = -1e30f, e[10], d = 0.f;
    for (int u = 0; u < 10; ++u) m = fmaxf(m, s_att[ch][u][v]);
    for (int u = 0; u < 10; ++u) { e[u] = __expf(s_att[ch][u][v] - m); d += e[u]; }
    const float rd = 1.f / d;
    for (int u = 0; u < 10; ++u) s_att[ch][u][v] = e[u] * rd;
  }
  __syncthreads();

  // ---- stage 4: reload raw x as f16 -> s_R1 (LN values dead) ----
  {
    const int w = tid;
    for (int c = 0; c < CCH; ++c) {
      s_R1[0][w * CCH + c] = (_Float16)rowl[(size_t)c * PLANE + w];
      s_R1[1][w * CCH + c] = (_Float16)rowr[(size_t)c * PLANE + w];
    }
  }
  __syncthreads();

  // ---- stage 5: V = P2 @ x + b2  (WMMA) -> s_R2 (Q dead) ----
  {
    const int wave = tid >> 5, lane = tid & 31;
    const int lhi = lane >> 4, llo = lane & 15;
    for (int i = 0; i < 8; ++i) {
      const int t8 = wave * 8 + i;
      const int mt = t8 & 3, nt = t8 >> 2;
      #pragma unroll
      for (int side = 0; side < 2; ++side) {
        v8f acc;
        #pragma unroll
        for (int r = 0; r < 8; ++r) acc[r] = s_vec[6 + side][mt * 16 + lhi * 8 + r];
        acc = mm_tile(s_w[2 + side], s_R1[side], mt, nt, lane, acc);
        #pragma unroll
        for (int r = 0; r < 8; ++r)
          s_R2[side][(nt * 16 + llo) * CCH + mt * 16 + lhi * 8 + r] = (_Float16)acc[r];
      }
    }
  }
  __syncthreads();

  // ---- stage 6: F_r2l = A_r2l @ Vr * beta ; F_l2r = A_l2r^T app. Vl * gamma -> s_R1 ----
  {
    const int w = tid, ch = tid / 10, u = tid % 10;
    float ar[10], al[10];
    #pragma unroll
    for (int v = 0; v < 10; ++v) ar[v] = s_ar[ch][u][v];        // row u of A_r2l
    #pragma unroll
    for (int uu = 0; uu < 10; ++uu) al[uu] = s_att[ch][uu][u];  // col u of A_l2r
    for (int c = 0; c < CCH; ++c) {
      float fl = 0.f, fr = 0.f;
      #pragma unroll
      for (int v = 0; v < 10; ++v) {
        fl += ar[v] * (float)s_R2[1][(ch * 10 + v) * CCH + c];  // Vr
        fr += al[v] * (float)s_R2[0][(ch * 10 + v) * CCH + c];  // Vl
      }
      s_R1[0][w * CCH + c] = (_Float16)(fl * s_vec[8][c]);      // * beta
      s_R1[1][w * CCH + c] = (_Float16)(fr * s_vec[9][c]);      // * gamma
    }
  }
  __syncthreads();

  // ---- stage 7: dynamic conv (WMMA) + bagg + residual -> global out ----
  {
    const int wave = tid >> 5, lane = tid & 31;
    const int lhi = lane >> 4, llo = lane & 15;
    float* outl = out + (size_t)b * IMG + (size_t)h * WW + wofs;
    float* outr = out + OUTR_OFS + (size_t)b * IMG + (size_t)h * WW + wofs;
    for (int i = 0; i < 8; ++i) {
      const int t8 = wave * 8 + i;
      const int mt = t8 & 3, nt = t8 >> 2;
      #pragma unroll
      for (int side = 0; side < 2; ++side) {
        v8f acc;
        #pragma unroll
        for (int r = 0; r < 8; ++r) acc[r] = s_vec[10 + side][mt * 16 + lhi * 8 + r];
        acc = mm_tile(s_w[4 + side], s_R1[side], mt, nt, lane, acc);
        const float* rx = side ? rowr : rowl;
        float* po = side ? outr : outl;
        #pragma unroll
        for (int r = 0; r < 8; ++r) {
          const int m_   = mt * 16 + lhi * 8 + r;
          const int col  = nt * 16 + llo;
          po[(size_t)m_ * PLANE + col] = acc[r] + rx[(size_t)m_ * PLANE + col];
        }
      }
    }
  }
}

// ---------------------------------------------------------------------------
extern "C" void kernel_launch(void* const* d_in, const int* in_sizes, int n_in,
                              void* d_out, int out_size, void* d_ws, size_t ws_size,
                              hipStream_t stream) {
  const float* xl   = (const float*)d_in[0];
  const float* xr   = (const float*)d_in[1];
  const float* nlw  = (const float*)d_in[2];
  const float* nlb  = (const float*)d_in[3];
  const float* nrw  = (const float*)d_in[4];
  const float* nrb  = (const float*)d_in[5];
  const float* lp1w = (const float*)d_in[6];
  const float* lp1b = (const float*)d_in[7];
  const float* rp1w = (const float*)d_in[8];
  const float* rp1b = (const float*)d_in[9];
  const float* lp2w = (const float*)d_in[10];
  const float* lp2b = (const float*)d_in[11];
  const float* rp2w = (const float*)d_in[12];
  const float* rp2b = (const float*)d_in[13];
  const float* beta  = (const float*)d_in[14];
  const float* gamma = (const float*)d_in[15];
  const float* lf1w = (const float*)d_in[16];
  const float* lf1b = (const float*)d_in[17];
  const float* lf2w = (const float*)d_in[18];
  const float* lf2b = (const float*)d_in[19];
  const float* lfW  = (const float*)d_in[20];
  const float* lfb  = (const float*)d_in[21];
  const float* rf1w = (const float*)d_in[22];
  const float* rf1b = (const float*)d_in[23];
  const float* rf2w = (const float*)d_in[24];
  const float* rf2b = (const float*)d_in[25];
  const float* rfW  = (const float*)d_in[26];
  const float* rfb  = (const float*)d_in[27];
  float* ws  = (float*)d_ws;
  float* out = (float*)d_out;

  gap_kernel<<<512, 256, 0, stream>>>(xl, xr, ws);
  attw_kernel<<<8, 64, 0, stream>>>(lf1w, lf1b, lf2w, lf2b, lfW, lfb,
                                    rf1w, rf1b, rf2w, rf2b, rfW, rfb, ws);
  fused_kernel<<<BB * HH * 2, 160, 0, stream>>>(
      xl, xr, nlw, nlb, nrw, nrb, lp1w, lp1b, rp1w, rp1b,
      lp2w, lp2b, rp2w, rp2b, beta, gamma, ws, out);
}
